// ModulatedConv2d_71854802862290
// MI455X (gfx1250) — matrix operand order repaired and checked by reading, not compiled, not run on hardware
//
#include <hip/hip_runtime.h>
#include <hip/hip_bf16.h>

// ---------------------------------------------------------------------------
// ModulatedConv2d (StyleGAN2) for gfx1250:
//   1) style_kernel : s = style @ mod_w^T + mod_b                 [8,256]
//   2) modw_kernel  : w = SCALE*weight*s ; demod ; bf16 [b][o][tap][i]
//   3) xcvt_kernel  : x NCHW f32 -> NHWC bf16 (one-time convert+transpose)
//   4) conv_kernel  : implicit GEMM (9 tap-GEMMs), double-buffered LDS,
//                     load/store-split software pipeline,
//                     v_wmma_f32_16x16x32_bf16, + LeakyReLU(0.2)
// ---------------------------------------------------------------------------

typedef __attribute__((ext_vector_type(16))) __bf16       v16bf;
typedef __attribute__((ext_vector_type(8)))  float        v8f;
typedef __attribute__((ext_vector_type(4)))  unsigned int uint4v;

#define IN_C   256
#define OUT_C  256
#define HW     128
#define SD     512
#define NB     8
#define SCALE  (1.0f / 48.0f)     // 1/sqrt(256*9)
#define NSLOPE 0.2f

#define LROW   80                 // LDS row stride bytes (32 bf16 padded to 80B)

union ABFrag { v16bf v; uint4v u[2]; };

static __device__ inline unsigned short f2bf(float f) {
    unsigned int u = __float_as_uint(f);
    unsigned int r = (u + 0x7fffu + ((u >> 16) & 1u)) >> 16;  // RNE
    return (unsigned short)r;
}

// -------------------------------- kernel 1 ---------------------------------
__global__ __launch_bounds__(256) void style_kernel(
    const float* __restrict__ style, const float* __restrict__ mod_w,
    const float* __restrict__ mod_b, float* __restrict__ s)
{
    const int b = blockIdx.x;
    const int i = threadIdx.x;
    const float* st = style + b * SD;
    const float* mw = mod_w + i * SD;
    float acc = 0.f;
#pragma unroll 8
    for (int k = 0; k < SD; ++k) acc += st[k] * mw[k];
    s[b * IN_C + i] = acc + mod_b[i];
}

// -------------------------------- kernel 2 ---------------------------------
__global__ __launch_bounds__(256) void modw_kernel(
    const float* __restrict__ weight, const float* __restrict__ s,
    unsigned short* __restrict__ wmod)
{
    const int o = blockIdx.x;
    const int b = blockIdx.y;
    const int i = threadIdx.x;

    const float* wrow = weight + (size_t)(o * IN_C + i) * 9;
    const float  sv   = s[b * IN_C + i];

    float vals[9];
    float ss = 0.f;
#pragma unroll
    for (int t = 0; t < 9; ++t) {
        float v = SCALE * wrow[t] * sv;
        vals[t] = v;
        ss += v * v;
    }
    __shared__ float red[256];
    red[i] = ss;
    __syncthreads();
#pragma unroll
    for (int off = 128; off > 0; off >>= 1) {
        if (i < off) red[i] += red[i + off];
        __syncthreads();
    }
    const float demod = rsqrtf(red[0] + 1e-8f);

    unsigned short* out = wmod + ((size_t)(b * OUT_C + o) * 9) * IN_C + i;
#pragma unroll
    for (int t = 0; t < 9; ++t) out[t * IN_C] = f2bf(vals[t] * demod);
}

// -------------------------------- kernel 3 ---------------------------------
// NCHW f32 -> NHWC bf16 : x_t[b][h][w][c]; grid (h=128, b=8), 256 threads
__global__ __launch_bounds__(256) void xcvt_kernel(
    const float* __restrict__ x, unsigned short* __restrict__ xt)
{
    __shared__ alignas(16) unsigned char tile[HW * LROW];   // 10 KB
    const int hrow = blockIdx.x;
    const int b    = blockIdx.y;
    const int tid  = threadIdx.x;

    for (int cc = 0; cc < 8; ++cc) {          // 32-channel chunks
        __syncthreads();
        // gather: coalesced reads along w, convert, transpose into LDS[w][c]
#pragma unroll
        for (int j = 0; j < 8; ++j) {
            const int idx   = j * 256 + tid;
            const int cpair = idx >> 7;       // 0..15
            const int w     = idx & 127;
            const int c0    = cpair * 2;
            const float* xp = x +
                (((size_t)(b * IN_C + cc * 32 + c0) * HW + hrow) << 7) + w;
            unsigned int packed =
                (unsigned int)f2bf(xp[0]) |
                ((unsigned int)f2bf(xp[HW * HW]) << 16);
            *(unsigned int*)(tile + w * LROW + c0 * 2) = packed;
        }
        __syncthreads();
        // scatter: contiguous channel runs per pixel -> b128 global stores
        const int w     = tid >> 1;
        const int chalf = (tid & 1) * 16;
        const unsigned char* p = tile + w * LROW + chalf * 2;
        uint4v v0 = *(const uint4v*)p;
        uint4v v1 = *(const uint4v*)(p + 16);
        unsigned short* op = xt +
            (((size_t)(b * HW + hrow) << 7) + w) * IN_C + cc * 32 + chalf;
        *(uint4v*)op       = v0;
        *(uint4v*)(op + 8) = v1;
    }
}

// -------------------------------- kernel 4 ---------------------------------
// grid: (h=128, m_tile=2, b=8); 256 threads = 8 waves (wave32)
// WG tile: 128 out-ch x 128 px (one image row). waves: 4(M) x 2(N),
// each wave 32x64 = 2x4 WMMA tiles. K: 9 taps x 8 chunks of 32 ch = 72 steps.
// Pipeline per step: barrier ; issue loads(s+1) ; WMMAs(s) ; commit LDS(s+1)
__global__ __launch_bounds__(256) void conv_kernel(
    const unsigned short* __restrict__ xt,   // NHWC bf16
    const unsigned short* __restrict__ wmod, // [b][o][tap][i] bf16
    float* __restrict__ out)
{
    __shared__ alignas(16) unsigned char ldsA[2][HW * LROW];  // 2 x 10 KB
    __shared__ alignas(16) unsigned char ldsB[2][HW * LROW];  // 2 x 10 KB

    const int h      = blockIdx.x;
    const int m_tile = blockIdx.y;        // 0..1 (128 out-ch each)
    const int b      = blockIdx.z;
    const int tid    = threadIdx.x;
    const int lane   = tid & 31;
    const int wave   = tid >> 5;
    const int wave_m = wave >> 1;         // 0..3
    const int wave_n = wave & 1;          // 0..1

    const v8f vzero = {0.f, 0.f, 0.f, 0.f, 0.f, 0.f, 0.f, 0.f};
    v8f acc[2][4] = {{vzero, vzero, vzero, vzero},
                     {vzero, vzero, vzero, vzero}};

    // staging indices: thread -> (row, 16-channel half)
    const int st_row   = tid >> 1;            // 0..127
    const int st_chalf = (tid & 1) * 16;      // 0 | 16 (bf16 units)

    // fragment addressing (contiguous k order: k = 16*(lane/16) + e, same for A and B)
    const bool lane_lo   = (lane < 16);
    const int  a_fr_row  = wave_m * 32 + (lane & 15);
    const unsigned a_off = lane_lo ? 0u : 32u;    // K 0..15 | 16..31 (+16 next)
    const int  b_fr_px   = wave_n * 64 + (lane & 15);
    const unsigned b_off = lane_lo ? 0u : 32u;    // K 0..15 | 16..31 (+16 next)

    struct StageRegs { uint4v a0, a1, b0, b1; };

    // issue global loads for K-step s (no LDS traffic, no waits needed here)
    auto stage_load = [&](int s) -> StageRegs {
        const int tap = s >> 3;               // 0..8
        const int ch  = s & 7;                // 0..7
        const int kh  = tap / 3;
        const int kw  = tap - kh * 3;
        StageRegs r;
        // A: wmod[b][m_tile*128 + st_row][tap][ch*32 + st_chalf ..+15]
        const unsigned short* ga = wmod +
            (((size_t)(b * OUT_C + m_tile * 128 + st_row) * 9 + tap) << 8) +
            ch * 32 + st_chalf;
        r.a0 = *(const uint4v*)ga;
        r.a1 = *(const uint4v*)(ga + 8);
        // B: x_t[b][h+kh-1][st_row+kw-1][ch*32 + st_chalf ..+15]
        const int src_h = h + kh - 1;
        const int src_w = st_row + kw - 1;
        uint4v z = {0u, 0u, 0u, 0u};
        r.b0 = z;
        r.b1 = z;
        if (((unsigned)src_h < (unsigned)HW) &&
            ((unsigned)src_w < (unsigned)HW)) {
            const unsigned short* gb = xt +
                (((size_t)(b * HW + src_h) << 7) + src_w) * IN_C +
                ch * 32 + st_chalf;
            r.b0 = *(const uint4v*)gb;
            r.b1 = *(const uint4v*)(gb + 8);
        }
        return r;
    };

    // commit staged registers into LDS buffer `buf`
    auto stage_store = [&](int buf, const StageRegs& r) {
        unsigned char* pa = ldsA[buf] + st_row * LROW + st_chalf * 2;
        *(uint4v*)pa        = r.a0;
        *(uint4v*)(pa + 16) = r.a1;
        unsigned char* pb = ldsB[buf] + st_row * LROW + st_chalf * 2;
        *(uint4v*)pb        = r.b0;
        *(uint4v*)(pb + 16) = r.b1;
    };

    auto compute = [&](int buf) {
        ABFrag afr[2], bfr[4];
#pragma unroll
        for (int mt = 0; mt < 2; ++mt) {
            const unsigned char* p =
                ldsA[buf] + (a_fr_row + mt * 16) * LROW + a_off;
            afr[mt].u[0] = *(const uint4v*)(p);
            afr[mt].u[1] = *(const uint4v*)(p + 16);
        }
#pragma unroll
        for (int nt = 0; nt < 4; ++nt) {
            const unsigned char* p =
                ldsB[buf] + (b_fr_px + nt * 16) * LROW + b_off;
            bfr[nt].u[0] = *(const uint4v*)(p);
            bfr[nt].u[1] = *(const uint4v*)(p + 16);
        }
#pragma unroll
        for (int mt = 0; mt < 2; ++mt)
#pragma unroll
            for (int nt = 0; nt < 4; ++nt)
                acc[mt][nt] = __builtin_amdgcn_wmma_f32_16x16x32_bf16(
                    false, afr[mt].v, false, bfr[nt].v,
                    (short)0, acc[mt][nt], false, false);
    };

    // -------- software-pipelined main loop: 72 K-steps, 1 barrier/step -----
    {
        StageRegs r0 = stage_load(0);
        stage_store(0, r0);
    }
    for (int s = 0; s < 72; ++s) {
        __syncthreads();
        StageRegs r;
        if (s < 71) r = stage_load(s + 1);   // loads fly during WMMAs below
        compute(s & 1);
        if (s < 71) stage_store((s + 1) & 1, r);
    }

    // epilogue: LeakyReLU + store (C layout: VGPR r -> M=r|r+8, lane -> N)
    const int n_col = lane & 15;
    const int m_add = lane_lo ? 0 : 8;
#pragma unroll
    for (int mt = 0; mt < 2; ++mt) {
#pragma unroll
        for (int nt = 0; nt < 4; ++nt) {
            const int o = m_tile * 128 + wave_m * 32 + mt * 16 + m_add;
            const int w = wave_n * 64 + nt * 16 + n_col;
            float* op = out + (((size_t)(b * OUT_C + o) * HW + h) << 7) + w;
#pragma unroll
            for (int r = 0; r < 8; ++r) {
                float v = acc[mt][nt][r];
                v = (v >= 0.f) ? v : NSLOPE * v;
                op[(size_t)r << 14] = v;   // +r output channels (128*128)
            }
        }
    }
}

// ---------------------------------------------------------------------------
extern "C" void kernel_launch(void* const* d_in, const int* in_sizes, int n_in,
                              void* d_out, int out_size, void* d_ws, size_t ws_size,
                              hipStream_t stream)
{
    const float* x      = (const float*)d_in[0];   // [8,256,128,128]
    const float* style  = (const float*)d_in[1];   // [8,512]
    const float* weight = (const float*)d_in[2];   // [256,256,3,3]
    const float* mod_w  = (const float*)d_in[3];   // [256,512]
    const float* mod_b  = (const float*)d_in[4];   // [256]
    float*       out    = (float*)d_out;           // [8,256,128,128]

    char* ws = (char*)d_ws;
    float*          s    = (float*)ws;                               // 8 KB
    unsigned short* wmod = (unsigned short*)(ws + 8192);             // 9.4 MB
    unsigned short* xt   = (unsigned short*)(ws + 8192 + 9437184);   // 67 MB

    style_kernel<<<dim3(NB), dim3(256), 0, stream>>>(style, mod_w, mod_b, s);
    modw_kernel<<<dim3(OUT_C, NB), dim3(256), 0, stream>>>(weight, s, wmod);
    xcvt_kernel<<<dim3(HW, NB), dim3(256), 0, stream>>>(x, xt);
    conv_kernel<<<dim3(HW, 2, NB), dim3(256), 0, stream>>>(xt, wmod, out);
}
